// MultiheadAttention_15607911153822
// MI455X (gfx1250) — compile-verified, matrix operand order checked by reading, not run on hardware
//
#include <hip/hip_runtime.h>

// MultiheadAttention forward for MI455X (gfx1250, wave32, WMMA, TDM).
// Pipeline: 3x bf16-WMMA GEMM (Q/K/V proj) -> RoPE -> flash attention
//           (TDM K-tile loads, bf16 WMMA, f32 online softmax with DPP
//           reductions) -> bf16-WMMA GEMM.

#define BATCH   4
#define S_LEN   2048
#define D_MODEL 1024
#define NHEADS  16
#define DK      64
#define M_TOT   (BATCH * S_LEN)   // 8192 rows

typedef __bf16 bf16;
typedef __attribute__((ext_vector_type(16))) __bf16 v16bf;
typedef __attribute__((ext_vector_type(8)))  __bf16 bf16x8;
typedef __attribute__((ext_vector_type(8)))  float  v8f;
typedef __attribute__((ext_vector_type(4)))  unsigned int u32x4;
typedef __attribute__((ext_vector_type(8)))  unsigned int u32x8;

// ---------------------------------------------------------------------------
// WMMA 16x32 bf16 fragment loader. Operand stored row-major with the WMMA
// K-dimension contiguous per row (stride `ld` elements). Per ISA layout:
//   lanes 0-15 : M/N = lane,    elems 0-7 -> K=k0+0..7,  elems 8-15 -> K=k0+16..23
//   lanes 16-31: M/N = lane-16, elems 0-7 -> K=k0+8..15, elems 8-15 -> K=k0+24..31
// All call sites keep (row*ld + k) 16B aligned so these lower to b128 loads.
// ---------------------------------------------------------------------------
__device__ __forceinline__ v16bf load_frag(const bf16* base, int rowbase, int ld, int k0) {
  const int lane = threadIdx.x & 31;
  const int row  = rowbase + (lane & 15);
  const int kb   = k0 + ((lane & 16) ? 8 : 0);
  const bf16* p  = base + row * ld + kb;
  bf16x8 lo = *(const bf16x8*)(p);
  bf16x8 hi = *(const bf16x8*)(p + 16);
  v16bf r;
#pragma unroll
  for (int i = 0; i < 8; ++i) { r[i] = lo[i]; r[i + 8] = hi[i]; }
  return r;
}

// ---------------------------------------------------------------------------
// VALU-only xor-style lane swaps via DPP16 (avoids ds_bpermute: keeps the LDS
// pipe free for WMMA fragment loads / TDM). Reductions act within each
// 16-lane half, matching the WMMA C-tile row layout.
//   xor1 -> quad_perm(1,0,3,2)=0xB1, xor2 -> quad_perm(2,3,0,1)=0x4E,
//   xor4 -> row_half_mirror=0x141,   xor8 -> row_mirror=0x140
// ---------------------------------------------------------------------------
template <int CTRL>
__device__ __forceinline__ float dpp_swap(float x) {
  int i = __float_as_int(x);
  i = __builtin_amdgcn_update_dpp(0, i, CTRL, 0xF, 0xF, true);
  return __int_as_float(i);
}
__device__ __forceinline__ float grp_max(float v) {
  v = fmaxf(v, dpp_swap<0xB1>(v));
  v = fmaxf(v, dpp_swap<0x4E>(v));
  v = fmaxf(v, dpp_swap<0x141>(v));
  v = fmaxf(v, dpp_swap<0x140>(v));
  return v;
}
__device__ __forceinline__ float grp_sum(float v) {
  v += dpp_swap<0xB1>(v);
  v += dpp_swap<0x4E>(v);
  v += dpp_swap<0x141>(v);
  v += dpp_swap<0x140>(v);
  return v;
}

// ---------------------------------------------------------------------------
// GEMM: C[m,n] = sum_k A[m,k] * W[n,k]   (A: MxK f32, W: NxK f32)
// Block tile 128x128, K-step 64, 256 threads = 8 waves (4x2); each wave owns a
// 32x64 output tile -> 8 WMMAs per 6 fragment loads per 32-deep k-chunk.
// HEADLAYOUT=1: store bf16 scattered to [B,H,S,64]; else f32 row-major.
// ---------------------------------------------------------------------------
constexpr int BM = 128, BN = 128, BK = 64;
constexpr int LDT = BK + 8;  // 72 elems -> 144B rows: b128-aligned, no bank conflicts

template <int HEADLAYOUT>
__global__ __launch_bounds__(256) void gemm_wmma(
    const float* __restrict__ A, const float* __restrict__ W,
    float* __restrict__ outF, bf16* __restrict__ outB)
{
  __shared__ __align__(16) bf16 As[BM * LDT];
  __shared__ __align__(16) bf16 Bs[BN * LDT];

  const int tid  = threadIdx.x;
  const int lane = tid & 31;
  const int wid  = tid >> 5;
  const int wm   = (wid >> 1) * 32;   // wave M offset in block tile
  const int wn   = (wid & 1) * 64;    // wave N offset
  const int m0   = blockIdx.y * BM;
  const int n0   = blockIdx.x * BN;

  v8f acc[2][4] = {};

  const int ldr = tid >> 4;        // 0..15  (tile-load row)
  const int ldc = (tid & 15) * 4;  // 0..60  (tile-load col, float4 granules)

  for (int k0 = 0; k0 < D_MODEL; k0 += BK) {
    __syncthreads();
#pragma unroll
    for (int rep = 0; rep < 8; ++rep) {
      int row = rep * 16 + ldr;
      float4 av = *(const float4*)(A + (size_t)(m0 + row) * D_MODEL + k0 + ldc);
      float4 wv = *(const float4*)(W + (size_t)(n0 + row) * D_MODEL + k0 + ldc);
      bf16* ap = &As[row * LDT + ldc];
      bf16* wp = &Bs[row * LDT + ldc];
      ap[0] = (bf16)av.x; ap[1] = (bf16)av.y; ap[2] = (bf16)av.z; ap[3] = (bf16)av.w;
      wp[0] = (bf16)wv.x; wp[1] = (bf16)wv.y; wp[2] = (bf16)wv.z; wp[3] = (bf16)wv.w;
    }
    // prefetch next k-slab (lowers to global_prefetch_b8); wave-uniform branch
    if (k0 + BK < D_MODEL) {
      int r2 = tid & 127;
      const float* p = (tid < 128)
          ? A + (size_t)(m0 + r2) * D_MODEL + k0 + BK
          : W + (size_t)(n0 + r2) * D_MODEL + k0 + BK;
      __builtin_prefetch(p, 0, 1);
    }
    __syncthreads();
#pragma unroll
    for (int kk = 0; kk < BK; kk += 32) {
      v16bf a0 = load_frag(As, wm,      LDT, kk);
      v16bf a1 = load_frag(As, wm + 16, LDT, kk);
#pragma unroll
      for (int j = 0; j < 4; ++j) {
        v16bf b = load_frag(Bs, wn + j * 16, LDT, kk);
        acc[0][j] = __builtin_amdgcn_wmma_f32_16x16x32_bf16(false, a0, false, b, (short)0, acc[0][j], false, false);
        acc[1][j] = __builtin_amdgcn_wmma_f32_16x16x32_bf16(false, a1, false, b, (short)0, acc[1][j], false, false);
      }
    }
  }

  // C tile layout: VGPR r -> row r (lanes 0-15) / row r+8 (lanes 16-31), col = lane%16
  const int nlo = lane & 15;
  const int mhi = (lane & 16) ? 8 : 0;
#pragma unroll
  for (int i = 0; i < 2; ++i) {
#pragma unroll
    for (int j = 0; j < 4; ++j) {
      int ncol = n0 + wn + j * 16 + nlo;
#pragma unroll
      for (int r = 0; r < 8; ++r) {
        int mrow = m0 + wm + i * 16 + r + mhi;
        float v = acc[i][j][r];
        if (HEADLAYOUT) {
          int b = mrow >> 11, s = mrow & (S_LEN - 1);
          int h = ncol >> 6,  d = ncol & (DK - 1);
          outB[((((size_t)b * NHEADS + h) * S_LEN + s) << 6) + d] = (bf16)v;
        } else {
          outF[(size_t)mrow * D_MODEL + ncol] = v;
        }
      }
    }
  }
}

// ---------------------------------------------------------------------------
// RoPE applied in place to Q and K (bf16, [B,H,S,64]). One thread per even/odd
// pair. Bandwidth-trivial (64 MB r/w vs 23.3 TB/s).
// ---------------------------------------------------------------------------
__global__ __launch_bounds__(256) void rope_qk(bf16* __restrict__ Q, bf16* __restrict__ K,
                                               const int* __restrict__ pos)
{
  int idx = blockIdx.x * blockDim.x + threadIdx.x;   // B*H*S*32 exactly
  int i  = idx & 31;                // pair index within head
  int s  = (idx >> 5) & (S_LEN - 1);
  int bh = idx >> 16;
  float p    = (float)pos[s];
  float freq = __expf(-(float)(2 * i) * (9.210340371976184f / 64.0f)); // theta^(-2i/dk)
  float sn, cs;
  __sincosf(p * freq, &sn, &cs);
  size_t base = (((size_t)bh * S_LEN + s) << 6) + 2 * i;
  float q0 = (float)Q[base], q1 = (float)Q[base + 1];
  Q[base]     = (bf16)(cs * q0 - sn * q1);
  Q[base + 1] = (bf16)(sn * q0 + cs * q1);
  float k0 = (float)K[base], k1 = (float)K[base + 1];
  K[base]     = (bf16)(cs * k0 - sn * k1);
  K[base + 1] = (bf16)(sn * k0 + cs * k1);
}

// ---------------------------------------------------------------------------
// Flash attention. Block = (b,h, 64 query rows); 4 waves x 16 rows each.
// K tile (32x64 bf16) is fetched by the Tensor Data Mover into padded LDS
// (pad_interval=32 DWORDs, pad_amount=4 DWORDs -> 72-element row stride),
// fenced with s_wait_tensorcnt. V is staged transposed by VALU (TDM has no
// transpose). Scores via 4 WMMAs, online softmax in f32 with DPP reductions,
// P -> A-fragment layout via per-wave LDS scratch, P*V via 4 WMMAs.
// ---------------------------------------------------------------------------
constexpr int LDKT = DK + 8;   // 72: K tile rows (32 keys x 64 d, d contiguous)
constexpr int LDVT = 32 + 8;   // 40: V^T rows  (64 d  x 32 keys, keys contiguous)
constexpr int LDPT = 32 + 8;   // 40: P scratch (16 q  x 32 keys)

__global__ __launch_bounds__(128) void attn_fwd(
    const bf16* __restrict__ Q, const bf16* __restrict__ K,
    const bf16* __restrict__ V, float* __restrict__ O)
{
  __shared__ __align__(16) bf16 Kt[32 * LDKT];
  __shared__ __align__(16) bf16 Vt[DK * LDVT];
  __shared__ __align__(16) bf16 Ps[4 * 16 * LDPT];

  const int tid  = threadIdx.x;
  const int lane = tid & 31;
  const int wid  = tid >> 5;
  const int lrw  = lane & 15;
  const int mhi  = (lane & 16) ? 8 : 0;

  const int bh = blockIdx.y;
  const int q0 = blockIdx.x * 64;
  const int qw = q0 + wid * 16;

  const bf16* Qh = Q + (size_t)bh * S_LEN * DK;
  const bf16* Kh = K + (size_t)bh * S_LEN * DK;
  const bf16* Vh = V + (size_t)bh * S_LEN * DK;

  // Q A-fragments stay resident (16 rows x 64 d = 2 fragments)
  v16bf qf[2];
#pragma unroll
  for (int kk = 0; kk < 2; ++kk) qf[kk] = load_frag(Qh, qw, DK, kk * 32);

  float mrow[8], lsum[8];
  v8f o[4] = {};
#pragma unroll
  for (int r = 0; r < 8; ++r) { mrow[r] = -1e30f; lsum[r] = 0.0f; }

  bf16* Pw = Ps + wid * 16 * LDPT;
  const unsigned int ldsK = (unsigned int)(uintptr_t)(void*)Kt;  // LDS byte offset
  const int nkt = (q0 >> 5) + 2;          // causal: keys up to q0+63

  for (int kt = 0; kt < nkt; ++kt) {
    __syncthreads();
    if (wid == 0) {
      // --- Tensor Data Mover: 2D tile load Kh[kt*32 .. +32)[0..64) -> Kt ---
      // D# group0: count=1 | lds_addr | global_addr(57b) | type=2
      unsigned long long ga = (unsigned long long)(uintptr_t)(Kh + (size_t)kt * 32 * DK);
      u32x4 g0;
      g0[0] = 1u;                                   // count=1, user mode, no gather
      g0[1] = ldsK;                                 // LDS dest (bytes)
      g0[2] = (unsigned int)ga;                     // global_addr[31:0]
      g0[3] = ((unsigned int)(ga >> 32) & 0x1FFFFFFu) | (2u << 30);  // addr[56:32] | type=2
      // D# group1: data_size=2B | pad_enable | pad_interval=32dw | pad_amount=4dw
      //            tensor_dim0=64, tensor_dim1=2048, tile 64x32, dim0_stride=64
      u32x8 g1;
      g1[0] = (1u << 16) | (1u << 20) | (4u << 22) | (3u << 25);
      g1[1] = (unsigned int)DK << 16;               // tensor_dim0[15:0] (=64)
      g1[2] = ((unsigned int)S_LEN & 0xFFFFu) << 16;// tensor_dim0[31:16]=0 | tensor_dim1 lo
      g1[3] = (unsigned int)DK << 16;               // tensor_dim1 hi=0 | tile_dim0=64
      g1[4] = 32u;                                  // tile_dim1=32 | tile_dim2=0
      g1[5] = (unsigned int)DK;                     // tensor_dim0_stride lo (=64)
      g1[6] = 0u;                                   // stride hi | dim1_stride lo
      g1[7] = 0u;
      asm volatile("tensor_load_to_lds %0, %1" :: "s"(g0), "s"(g1) : "memory");
      __builtin_amdgcn_s_wait_tensorcnt(0);
    }
    {   // cooperative transposed V stage: Vt[d][j] = V[kt*32+j][d]
      int r = tid >> 2;
      int c = (tid & 3) * 16;
      const bf16* vsrc = Vh + (size_t)(kt * 32 + r) * DK + c;
#pragma unroll
      for (int i = 0; i < 16; ++i) Vt[(c + i) * LDVT + r] = vsrc[i];
      if (kt + 1 < nkt) __builtin_prefetch(vsrc + 32 * DK, 0, 1);
    }
    __syncthreads();

    // scores S = Q * K^T : 16 q x 32 keys
    v8f sc[2];
#pragma unroll
    for (int t = 0; t < 2; ++t) {
      v8f c = {};
#pragma unroll
      for (int kk = 0; kk < 2; ++kk) {
        v16bf b = load_frag(Kt, t * 16, LDKT, kk * 32);
        c = __builtin_amdgcn_wmma_f32_16x16x32_bf16(false, qf[kk], false, b, (short)0, c, false, false);
      }
      sc[t] = c;
    }

    // online softmax update + stage P in A-fragment layout
#pragma unroll
    for (int r = 0; r < 8; ++r) {
      int qabs = qw + r + mhi;
#pragma unroll
      for (int t = 0; t < 2; ++t) {
        int j = kt * 32 + t * 16 + lrw;
        float v = sc[t][r] * 0.125f;          // 1/sqrt(64)
        sc[t][r] = (j <= qabs) ? v : -1e30f;  // causal mask
      }
      float mx   = grp_max(fmaxf(sc[0][r], sc[1][r]));
      float mnew = fmaxf(mrow[r], mx);
      float alpha = __expf(mrow[r] - mnew);
      mrow[r] = mnew;
      float p0 = __expf(sc[0][r] - mnew);
      float p1 = __expf(sc[1][r] - mnew);
      lsum[r] = lsum[r] * alpha + grp_sum(p0 + p1);
#pragma unroll
      for (int n = 0; n < 4; ++n) o[n][r] *= alpha;
      Pw[(r + mhi) * LDPT + lrw]      = (bf16)p0;
      Pw[(r + mhi) * LDPT + 16 + lrw] = (bf16)p1;
    }
    __syncthreads();

    // O += P * V
    v16bf pa = load_frag(Pw, 0, LDPT, 0);
#pragma unroll
    for (int n = 0; n < 4; ++n) {
      v16bf bv = load_frag(Vt, n * 16, LDVT, 0);
      o[n] = __builtin_amdgcn_wmma_f32_16x16x32_bf16(false, pa, false, bv, (short)0, o[n], false, false);
    }
  }

  // normalize and write O as f32 [B, S, H*64] for the output projection
  const int b = bh >> 4, h = bh & 15;
#pragma unroll
  for (int n = 0; n < 4; ++n) {
#pragma unroll
    for (int r = 0; r < 8; ++r) {
      int s = qw + r + mhi;
      O[((size_t)(b * S_LEN + s)) * D_MODEL + h * DK + n * 16 + lrw] = o[n][r] / lsum[r];
    }
  }
}

// ---------------------------------------------------------------------------
extern "C" void kernel_launch(void* const* d_in, const int* in_sizes, int n_in,
                              void* d_out, int out_size, void* d_ws, size_t ws_size,
                              hipStream_t stream) {
  (void)in_sizes; (void)n_in; (void)out_size; (void)ws_size;
  const float* x   = (const float*)d_in[0];
  const float* Wq  = (const float*)d_in[1];
  const float* Wk  = (const float*)d_in[2];
  const float* Wv  = (const float*)d_in[3];
  const float* Wo  = (const float*)d_in[4];
  const int*   pos = (const int*)d_in[5];
  float* out = (float*)d_out;

  const size_t qkv = (size_t)BATCH * S_LEN * D_MODEL;   // 8M elements
  bf16*  Qw = (bf16*)d_ws;            // 16 MB each, bf16 [B,H,S,64]
  bf16*  Kw = Qw + qkv;
  bf16*  Vw = Kw + qkv;
  float* Ow = (float*)(Vw + qkv);     // 32 MB f32 [B,S,1024]

  dim3 ggrid(D_MODEL / BN, M_TOT / BM);                 // 8 x 64
  gemm_wmma<1><<<ggrid, 256, 0, stream>>>(x, Wq, nullptr, Qw);
  gemm_wmma<1><<<ggrid, 256, 0, stream>>>(x, Wk, nullptr, Kw);
  gemm_wmma<1><<<ggrid, 256, 0, stream>>>(x, Wv, nullptr, Vw);

  int ropeN = BATCH * NHEADS * S_LEN * 32;              // exact multiple of 256
  rope_qk<<<ropeN / 256, 256, 0, stream>>>(Qw, Kw, pos);

  dim3 agrid(S_LEN / 64, BATCH * NHEADS);               // 32 x 64
  attn_fwd<<<agrid, 128, 0, stream>>>(Qw, Kw, Vw, Ow);

  gemm_wmma<0><<<ggrid, 256, 0, stream>>>(Ow, Wo, out, nullptr);
}
